// RCNN_16767552323964
// MI455X (gfx1250) — compile-verified, matrix operand order checked by reading, not run on hardware
//
#include <hip/hip_runtime.h>
#include <hip/hip_bf16.h>

typedef _Float16 v16h __attribute__((ext_vector_type(16)));
typedef _Float16 v8h  __attribute__((ext_vector_type(8)));
typedef float    v8f  __attribute__((ext_vector_type(8)));

#define S_LEN 512
#define BATCH 64
#define EMB   300
#define EPAD  320
#define HID   256
#define MDIM  512
#define CDIM  20
#define CATW  812
#define CATP  832
#define NROW  (S_LEN*BATCH)   // 32768

// ---------------------------------------------------------------------------
// WMMA fragment loader. Row-major f16 matrix, tile = 16 rows x 32 k.
// CDNA5 16-bit A layout: lane l holds row (l&15); halves h=0..7 are
// k = 8*(l>>4)+0..7, halves h=8..15 are k = 16+8*(l>>4)+0..7.
// Bt (weights stored [n][k] row-major) uses the identical pattern with
// n = l&15. Both resolve to two contiguous 16B loads per lane.
// ---------------------------------------------------------------------------
__device__ __forceinline__ v16h load_frag(const _Float16* p, int ld) {
  int lane = threadIdx.x & 31;
  const _Float16* q = p + (lane & 15) * ld + ((lane >> 4) << 3);
  v8h lo = *(const v8h*)(q);
  v8h hi = *(const v8h*)(q + 16);
  return __builtin_shufflevector(lo, hi, 0,1,2,3,4,5,6,7,8,9,10,11,12,13,14,15);
}

__device__ __forceinline__ v8f wmma_f16(v16h a, v16h b, v8f c) {
  // emits v_wmma_f32_16x16x32_f16 (codegen-confirmed builtin, 8 args)
  return __builtin_amdgcn_wmma_f32_16x16x32_f16(
      /*neg_a=*/false, a, /*neg_b=*/false, b,
      /*c_mod=*/(short)0, c, /*reuse_a=*/false, /*reuse_b=*/false);
}

__device__ __forceinline__ void atomicMaxF32(float* addr, float val) {
  unsigned* ua = (unsigned*)addr;
  unsigned old = *ua;
  while (true) {
    float f = __uint_as_float(old);
    if (f >= val) break;
    unsigned assumed = old;
    old = atomicCAS(ua, assumed, __float_as_uint(val));
    if (old == assumed) break;
  }
}

// ---------------------------------------------------------------------------
// Kernel 1: convert weights to f16 (zero-padded K), init ymax to -inf.
// ---------------------------------------------------------------------------
__global__ void prep_kernel(const float* Wsl, const float* Wsr,
                            const float* Wl,  const float* Wr,
                            const float* maxw,
                            _Float16* WslH, _Float16* WsrH,
                            _Float16* WlH,  _Float16* WrH,
                            _Float16* maxwH, float* ymax) {
  int tid = blockIdx.x * blockDim.x + threadIdx.x;
  int stride = gridDim.x * blockDim.x;
  for (int i = tid; i < HID * EPAD; i += stride) {
    int h = i / EPAD, e = i % EPAD;
    WslH[i] = (_Float16)((e < EMB) ? Wsl[h * EMB + e] : 0.f);
    WsrH[i] = (_Float16)((e < EMB) ? Wsr[h * EMB + e] : 0.f);
  }
  for (int i = tid; i < HID * HID; i += stride) {
    WlH[i] = (_Float16)Wl[i];
    WrH[i] = (_Float16)Wr[i];
  }
  for (int i = tid; i < MDIM * CATP; i += stride) {
    int m = i / CATP, c = i % CATP;
    maxwH[i] = (_Float16)((c < CATW) ? maxw[m * CATW + c] : 0.f);
  }
  for (int i = tid; i < BATCH * MDIM; i += stride) ymax[i] = -3.0e38f;
}

// ---------------------------------------------------------------------------
// Kernel 2: embedding gather -> embH (padded f16) and cat[:,256:556] + pad.
// ---------------------------------------------------------------------------
__global__ void gather_kernel(const int* inp, const float* table,
                              _Float16* embH, _Float16* catH) {
  int r = blockIdx.x;                         // r = s*64 + b
  int idx = inp[r];
  const float* src = table + (size_t)idx * EMB;
  _Float16* e  = embH + (size_t)r * EPAD;
  _Float16* cg = catH + (size_t)r * CATP;
  for (int i = threadIdx.x; i < EPAD; i += blockDim.x) {
    float v = (i < EMB) ? src[i] : 0.f;
    e[i] = (_Float16)v;
    if (i < EMB) cg[HID + i] = (_Float16)v;
  }
  for (int i = threadIdx.x; i < CATP - CATW; i += blockDim.x)
    cg[CATW + i] = (_Float16)0.f;
}

// ---------------------------------------------------------------------------
// Kernel 3: emb_ls/emb_rs = emb @ W^T + b.  One wave per 16x16 tile,
// both projections share the A fragment.  K = 320 (padded) -> 10 chunks.
// ---------------------------------------------------------------------------
__global__ void proj_kernel(const _Float16* embH,
                            const _Float16* WslH, const _Float16* WsrH,
                            const float* bsl, const float* bsr,
                            float* xls, float* xrs) {
  int wid  = threadIdx.x >> 5;
  int lane = threadIdx.x & 31;
  int job = blockIdx.x * 8 + wid;      // 0..32767
  int rt = job >> 4;                   // row tile 0..2047
  int ht = job & 15;                   // h tile   0..15
  const _Float16* Ab = embH + (size_t)(rt * 16) * EPAD;
  const _Float16* B1 = WslH + (ht * 16) * EPAD;
  const _Float16* B2 = WsrH + (ht * 16) * EPAD;
  v8f acc1 = {0,0,0,0,0,0,0,0};
  v8f acc2 = {0,0,0,0,0,0,0,0};
  for (int kk = 0; kk < EPAD / 32; ++kk) {
    v16h a  = load_frag(Ab + kk * 32, EPAD);
    v16h b1 = load_frag(B1 + kk * 32, EPAD);
    v16h b2 = load_frag(B2 + kk * 32, EPAD);
    acc1 = wmma_f16(a, b1, acc1);
    acc2 = wmma_f16(a, b2, acc2);
  }
  int mrow = 8 * (lane >> 4);
  int n = lane & 15;
  int h = ht * 16 + n;
  float b1v = bsl[h], b2v = bsr[h];
  for (int i = 0; i < 8; ++i) {
    int r = rt * 16 + mrow + i;
    xls[(size_t)r * HID + h] = acc1[i] + b1v;
    xrs[(size_t)r * HID + h] = acc2[i] + b2v;
  }
}

// ---------------------------------------------------------------------------
// Kernel 4: the two sequential scans.  grid=2 (block 0: left, block 1: right),
// 1024 threads = 32 waves.  State c (64x256 f16 = 32KB) lives in LDS; W is
// L2-resident.  Per step each wave does 2 output tiles x 8 K-chunks of WMMA.
// Results written to LDS (next state) and into the cat buffer (f16).
// ---------------------------------------------------------------------------
__global__ __launch_bounds__(1024) void scan_kernel(
    const float* xl, const float* xr,
    const _Float16* Wl, const _Float16* Wr,
    const float* bl, const float* br,
    const float* c0l, const float* c0r,
    _Float16* catH) {
  __shared__ __align__(16) _Float16 cbuf[BATCH][HID];

  int dir = blockIdx.x;
  const float*    x    = dir ? xr  : xl;
  const _Float16* W    = dir ? Wr  : Wl;
  const float*    bias = dir ? br  : bl;
  const float*    c0   = dir ? c0r : c0l;
  int colOff = dir ? (HID + EMB) : 0;     // 556 or 0

  int tid = threadIdx.x;
  for (int i = tid; i < BATCH * HID; i += 1024)
    cbuf[i / HID][i % HID] = (_Float16)c0[i % HID];
  __syncthreads();

  int wid = tid >> 5, lane = tid & 31;
  int mrow = 8 * (lane >> 4);
  int n = lane & 15;

  for (int step = 0; step < S_LEN; ++step) {
    int s = dir ? (S_LEN - 1 - step) : step;
    float vout[2][8];
    for (int t = 0; t < 2; ++t) {
      int tile = wid * 2 + t;           // 0..63
      int bt = tile >> 4;               // batch tile 0..3
      int ht = tile & 15;               // hidden tile 0..15
      const _Float16* Ab = &cbuf[bt * 16][0];
      const _Float16* Bb = W + (ht * 16) * HID;
      v8f acc = {0,0,0,0,0,0,0,0};
      for (int kk = 0; kk < HID / 32; ++kk) {
        v16h a = load_frag(Ab + kk * 32, HID);
        v16h b = load_frag(Bb + kk * 32, HID);
        acc = wmma_f16(a, b, acc);
      }
      int h = ht * 16 + n;
      float bv = bias[h];
      for (int i = 0; i < 8; ++i) {
        int b = bt * 16 + mrow + i;
        float pre = acc[i] + bv + x[((size_t)s * BATCH + b) * HID + h];
        vout[t][i] = tanhf(pre);
      }
    }
    __syncthreads();                    // everyone done reading old state
    for (int t = 0; t < 2; ++t) {
      int tile = wid * 2 + t;
      int bt = tile >> 4, ht = tile & 15;
      int h = ht * 16 + n;
      for (int i = 0; i < 8; ++i) {
        int b = bt * 16 + mrow + i;
        _Float16 hv = (_Float16)vout[t][i];
        cbuf[b][h] = hv;
        catH[((size_t)s * BATCH + b) * CATP + colOff + h] = hv;
      }
    }
    __syncthreads();                    // new state visible
  }
}

// ---------------------------------------------------------------------------
// Kernel 5: y = tanh(cat @ max_w^T + max_b), running max over s in regs,
// then float atomic-max into ymax[b][m].  grid=(32 mt, 4 bt, 4 sc), 8 waves,
// wave handles 16 consecutive s values.  K = 832 -> 26 chunks.
// ---------------------------------------------------------------------------
__global__ void maxpool_kernel(const _Float16* catH, const _Float16* maxwH,
                               const float* mb, float* ymax) {
  int wid = threadIdx.x >> 5, lane = threadIdx.x & 31;
  int mt = blockIdx.x;     // 0..31
  int bt = blockIdx.y;     // 0..3
  int sc = blockIdx.z;     // 0..3
  const _Float16* Bb = maxwH + (mt * 16) * CATP;
  int mrow = 8 * (lane >> 4);
  int n = lane & 15;
  int mcol = mt * 16 + n;
  float bv = mb[mcol];
  float vmax[8];
  for (int i = 0; i < 8; ++i) vmax[i] = -3.0e38f;
  for (int it = 0; it < 16; ++it) {
    int s = sc * 128 + wid * 16 + it;
    const _Float16* Ab = catH + (size_t)(s * BATCH + bt * 16) * CATP;
    v8f acc = {0,0,0,0,0,0,0,0};
    for (int kk = 0; kk < CATP / 32; ++kk) {
      v16h a = load_frag(Ab + kk * 32, CATP);
      v16h b = load_frag(Bb + kk * 32, CATP);
      acc = wmma_f16(a, b, acc);
    }
    for (int i = 0; i < 8; ++i)
      vmax[i] = fmaxf(vmax[i], tanhf(acc[i] + bv));
  }
  for (int i = 0; i < 8; ++i) {
    int b = bt * 16 + mrow + i;
    atomicMaxF32(&ymax[b * MDIM + mcol], vmax[i]);
  }
}

// ---------------------------------------------------------------------------
// Kernel 6: classifier head + log_softmax.  64 threads, one per batch row.
// ---------------------------------------------------------------------------
__global__ void head_kernel(const float* ymax, const float* docw,
                            const float* docb, float* out) {
  int b = threadIdx.x;
  const float* ym = ymax + b * MDIM;
  float logit[CDIM];
  float mx = -3.0e38f;
  for (int c = 0; c < CDIM; ++c) {
    float acc = docb[c];
    const float* w = docw + c * MDIM;
    for (int m = 0; m < MDIM; ++m) acc += ym[m] * w[m];
    logit[c] = acc;
    mx = fmaxf(mx, acc);
  }
  float sum = 0.f;
  for (int c = 0; c < CDIM; ++c) sum += expf(logit[c] - mx);
  float lse = mx + logf(sum);
  for (int c = 0; c < CDIM; ++c) out[b * CDIM + c] = logit[c] - lse;
}

// ---------------------------------------------------------------------------
extern "C" void kernel_launch(void* const* d_in, const int* in_sizes, int n_in,
                              void* d_out, int out_size, void* d_ws, size_t ws_size,
                              hipStream_t stream) {
  const int*   inp    = (const int*)  d_in[0];
  const float* table  = (const float*)d_in[1];
  const float* c_l0   = (const float*)d_in[2];
  const float* c_r0   = (const float*)d_in[3];
  const float* W_l_w  = (const float*)d_in[4];
  const float* W_l_b  = (const float*)d_in[5];
  const float* W_r_w  = (const float*)d_in[6];
  const float* W_r_b  = (const float*)d_in[7];
  const float* W_sl_w = (const float*)d_in[8];
  const float* W_sl_b = (const float*)d_in[9];
  const float* W_sr_w = (const float*)d_in[10];
  const float* W_sr_b = (const float*)d_in[11];
  const float* max_w  = (const float*)d_in[12];
  const float* max_b  = (const float*)d_in[13];
  const float* doc_w  = (const float*)d_in[14];
  const float* doc_b  = (const float*)d_in[15];

  char* ws = (char*)d_ws;
  size_t off = 0;
  auto alloc = [&](size_t bytes) {
    void* p = ws + off;
    off = (off + bytes + 255) & ~(size_t)255;
    return p;
  };
  _Float16* WslH  = (_Float16*)alloc((size_t)HID * EPAD * 2);
  _Float16* WsrH  = (_Float16*)alloc((size_t)HID * EPAD * 2);
  _Float16* WlH   = (_Float16*)alloc((size_t)HID * HID * 2);
  _Float16* WrH   = (_Float16*)alloc((size_t)HID * HID * 2);
  _Float16* maxwH = (_Float16*)alloc((size_t)MDIM * CATP * 2);
  _Float16* embH  = (_Float16*)alloc((size_t)NROW * EPAD * 2);
  _Float16* catH  = (_Float16*)alloc((size_t)NROW * CATP * 2);
  float*    xls   = (float*)   alloc((size_t)NROW * HID * 4);
  float*    xrs   = (float*)   alloc((size_t)NROW * HID * 4);
  float*    ymax  = (float*)   alloc((size_t)BATCH * MDIM * 4);
  (void)ws_size; (void)in_sizes; (void)n_in; (void)out_size;

  prep_kernel<<<512, 256, 0, stream>>>(W_sl_w, W_sr_w, W_l_w, W_r_w, max_w,
                                       WslH, WsrH, WlH, WrH, maxwH, ymax);
  gather_kernel<<<NROW, 128, 0, stream>>>(inp, table, embH, catH);
  proj_kernel<<<4096, 256, 0, stream>>>(embH, WslH, WsrH, W_sl_b, W_sr_b,
                                        xls, xrs);
  scan_kernel<<<2, 1024, 0, stream>>>(xls, xrs, WlH, WrH, W_l_b, W_r_b,
                                      c_l0, c_r0, catH);
  maxpool_kernel<<<dim3(32, 4, 4), 256, 0, stream>>>(catH, maxwH, max_b, ymax);
  head_kernel<<<1, 64, 0, stream>>>(ymax, doc_w, doc_b, (float*)d_out);
}
